// TCNEnsemble_36086315221350
// MI455X (gfx1250) — matrix operand order repaired and checked by reading, not compiled, not run on hardware
//
#include <hip/hip_runtime.h>
#include <hip/hip_bf16.h>
#include <math.h>

// ---------------------------------------------------------------------------
// Types for CDNA5 WMMA (wave32, 16x16x32 bf16 -> f32)
// ---------------------------------------------------------------------------
typedef __attribute__((ext_vector_type(16))) __bf16    v16bf;
typedef __attribute__((ext_vector_type(8)))  float     v8f;
typedef __attribute__((ext_vector_type(4)))  unsigned  v4u;

union BFrag { v16bf v; v4u q[2]; unsigned short u[16]; };

__device__ __forceinline__ unsigned short f2bf(float f) {
    union { float f; unsigned u; } v; v.f = f;
    unsigned u = v.u;
    u += 0x7FFFu + ((u >> 16) & 1u);           // round-to-nearest-even
    return (unsigned short)(u >> 16);
}
__device__ __forceinline__ float bf2f(unsigned short h) {
    union { unsigned u; float f; } v; v.u = ((unsigned)h) << 16;
    return v.f;
}

// Problem constants
#define TB   32      // batch
#define TT   1024    // time
#define TF   8       // features
#define CC   128     // channels
#define KK3  3       // kernel taps
#define CWIN 64      // pruned causal window (receptive field 61 <= 64)
#define NBAT 2       // batches per workgroup
#define NCOL (NBAT*CWIN)   // 128 GEMM columns per workgroup
#define NCONV_PER_F 7      // w2_0, then (w1_r,w2_r) x 3 levels
#define FRAGS_PER_CONV (3*4*8)   // taps * (128/32 ksteps) * (128/16 mtiles)
#define FRAG_ELEMS 512           // 32 lanes * 16 halves

// ---------------------------------------------------------------------------
// Kernel 1: RevIN stats + normalized last-64 window per (b,f)
// ---------------------------------------------------------------------------
__global__ __launch_bounds__(128) void tcn_stats_kernel(
    const float* __restrict__ x, const float* __restrict__ rw,
    const float* __restrict__ rb, float* __restrict__ xnw,
    float* __restrict__ ms)
{
    __shared__ float s1[128], s2[128];
    __shared__ float mean_s, inv_s, std_s;
    const int bf = blockIdx.x, b = bf >> 3, f = bf & 7;
    const int tid = threadIdx.x;
    float sum = 0.f, sq = 0.f;
    for (int r = 0; r < 8; ++r) {
        float v = x[((size_t)b*TT + tid + r*128)*TF + f];
        sum += v; sq += v*v;
    }
    s1[tid] = sum; s2[tid] = sq; __syncthreads();
    for (int st = 64; st > 0; st >>= 1) {
        if (tid < st) { s1[tid] += s1[tid+st]; s2[tid] += s2[tid+st]; }
        __syncthreads();
    }
    if (tid == 0) {
        float m  = s1[0] * (1.f/1024.f);
        float vr = s2[0] * (1.f/1024.f) - m*m;
        float sd = sqrtf(vr + 1e-5f);
        mean_s = m; std_s = sd; inv_s = 1.f/sd;
        if (f == 0) { ms[b] = m; ms[32 + b] = sd; }
    }
    __syncthreads();
    if (tid < CWIN) {
        int t = TT - CWIN + tid;
        float v = (x[((size_t)b*TT + t)*TF + f] - mean_s) * inv_s * rw[f] + rb[f];
        xnw[((size_t)f*TB + b)*CWIN + tid] = v;
    }
    (void)std_s;
}

// ---------------------------------------------------------------------------
// Kernel 2: swizzle conv weights f32 -> bf16 in exact A-fragment layout
//   A 16x32 bf16 order used here (contiguous k, same as B): lanes 0-15 hold M=lane, K={0..15};
//   lanes 16-31 hold M=lane-16, K={16..31}; 2 halves per VGPR.
// ---------------------------------------------------------------------------
__global__ __launch_bounds__(256) void tcn_wswz_kernel(
    const float* __restrict__ w2_0, const float* __restrict__ w1_r,
    const float* __restrict__ w2_r, unsigned short* __restrict__ wswz)
{
    const int idx = blockIdx.x * 256 + threadIdx.x;        // < 56*96*512
    const int i    = idx & 15;
    const int lane = (idx >> 4) & 31;
    const int frag = idx >> 9;
    const int mt   = frag & 7;
    const int kk   = (frag >> 3) & 3;
    const int tap  = (frag >> 5) % 3;
    const int cg   = frag / FRAGS_PER_CONV;                // 0..55
    const int f = cg / NCONV_PER_F, j = cg % NCONV_PER_F;
    const float* src;
    if (j == 0)      src = w2_0 + (size_t)f * CC*CC*KK3;
    else if (j & 1)  src = w1_r + (size_t)(((j-1)>>1)*TF + f) * CC*CC*KK3;
    else             src = w2_r + (size_t)(((j>>1)-1)*TF + f) * CC*CC*KK3;
    const int co = mt*16 + (lane & 15);
    const int kl = (lane < 16) ? i : i + 16;
    const int ci = kk*32 + kl;
    wswz[idx] = f2bf(src[((size_t)co*CC + ci)*KK3 + tap]);
}

// ---------------------------------------------------------------------------
// WMMA conv core: one 128->128 K=3 dilated causal conv over NCOL columns.
//   in/out LDS layout: H[n][ci] (n = 2 batches x 64 t), bf16.
//   MODE 0: leaky(acc+bias)
//   MODE 1: relu(leaky(acc+bias) + res[n][co])          (res == out, in-place)
//   MODE 2: relu(leaky(acc+bias) + wd[co]*xn[n]+bd[co]) (block0 downsample)
// ---------------------------------------------------------------------------
template <int MODE>
__device__ __forceinline__ void conv_wmma(
    const unsigned short* in, unsigned short* out,
    const unsigned short* __restrict__ wbase,
    const float* bias_l, int dil,
    const unsigned short* res,
    const float* xn_l, const float* wd_l, const float* bd_l,
    int lane, int wave)
{
    v8f acc[8] = {};
    const int mt   = wave;
    const int ncl  = lane & 15;
    const int hiK  = lane >> 4;                 // 0 or 1
    const int koff = hiK << 4;
    for (int tap = 0; tap < 3; ++tap) {
        const int s = (2 - tap) * dil;          // causal time shift
        for (int kk = 0; kk < 4; ++kk) {
            BFrag a;
            const v4u* ga = (const v4u*)(wbase +
                ((((tap << 2) + kk) * 8 + mt) << 9) + (lane << 4));
            a.q[0] = ga[0]; a.q[1] = ga[1];     // 2x global_load_b128
            __builtin_prefetch(ga + 64, 0, 1);  // global_prefetch_b8 ahead
            const int kbase = (kk << 5) + koff; // contiguous 16 ci per lane
            for (int nt = 0; nt < 8; ++nt) {
                const int n  = (nt << 4) + ncl;
                const int t  = n & 63;
                const int n1 = (t >= s) ? (n - s) : (n - t); // clamp in-batch
                BFrag b;
                const v4u* pb = (const v4u*)&in[n1 * CC + kbase];
                b.q[0] = pb[0]; b.q[1] = pb[1]; // 2x ds_load_b128
                acc[nt] = __builtin_amdgcn_wmma_f32_16x16x32_bf16(
                    false, a.v, false, b.v, (short)0, acc[nt], false, false);
            }
        }
    }
    // Epilogue: C/D 16x16 f32 layout -> H[n][co]
    const int co_base = mt*16 + (hiK << 3);
    for (int nt = 0; nt < 8; ++nt) {
        const int n = (nt << 4) + ncl;
        for (int j = 0; j < 8; ++j) {
            const int co = co_base + j;
            float v = acc[nt][j] + bias_l[co];
            v = v > 0.f ? v : 0.01f * v;                       // leaky_relu
            if (MODE == 1) v += bf2f(res[n * CC + co]);
            if (MODE == 2) v += wd_l[co] * xn_l[n] + bd_l[co];
            if (MODE != 0) v = v > 0.f ? v : 0.f;              // relu
            out[n * CC + co] = f2bf(v);
        }
    }
}

// ---------------------------------------------------------------------------
// Kernel 3: one TCN branch segment per workgroup: (feature, 2-batch group)
// ---------------------------------------------------------------------------
__global__ __launch_bounds__(256) void tcn_branch_kernel(
    const float* __restrict__ xnw, const unsigned short* __restrict__ wswz,
    const float* __restrict__ w1_0, const float* __restrict__ b1_0,
    const float* __restrict__ b2_0,
    const float* __restrict__ down_w, const float* __restrict__ down_b,
    const float* __restrict__ b1_r, const float* __restrict__ b2_r,
    float* __restrict__ feats)
{
    extern __shared__ unsigned char smem[];
    unsigned short* HA   = (unsigned short*)smem;        // [128 n][128 ci]
    unsigned short* HB   = HA + NCOL*CC;                 // [128 n][128 ci]
    float* xn_l   = (float*)(HB + NCOL*CC);              // [128]
    float* bias_l = xn_l + NCOL;                         // [128]
    float* wd_l   = bias_l + CC;                         // [128]
    float* bd_l   = wd_l + CC;                           // [128]

    const int f    = blockIdx.y;
    const int bg   = blockIdx.x;          // 0..15 -> batches {2bg, 2bg+1}
    const int tid  = threadIdx.x;
    const int lane = tid & 31;
    const int wave = tid >> 5;
    const unsigned short* wf = wswz +
        (size_t)(f * NCONV_PER_F) * FRAGS_PER_CONV * FRAG_ELEMS;

    if (tid < NCOL) {
        int bl = tid >> 6, t = tid & 63;
        xn_l[tid] = xnw[((size_t)f*TB + (bg*NBAT + bl))*CWIN + t];
    }
    if (tid < CC) {
        wd_l[tid] = down_w[f*CC + tid];
        bd_l[tid] = down_b[f*CC + tid];
    }
    __syncthreads();

    // ---- block 0, conv1 (1->C, dil=1): scalar, write HB ----
    for (int r = 0; r < 64; ++r) {
        int e = tid + (r << 8);
        int co = e >> 7, n = e & 127;
        int t = n & 63, base = n & ~63;
        float x0 = xn_l[base + (t >= 2 ? t-2 : 0)];
        float x1 = xn_l[base + (t >= 1 ? t-1 : 0)];
        float x2 = xn_l[n];
        const float* w = &w1_0[((size_t)f*CC + co)*KK3];
        float y = w[0]*x0 + w[1]*x1 + w[2]*x2 + b1_0[f*CC + co];
        HB[n*CC + co] = f2bf(y > 0.f ? y : 0.01f*y);
    }
    __syncthreads();

    // ---- block 0, conv2 (WMMA) + 1x1 downsample residual -> HA ----
    if (tid < CC) bias_l[tid] = b2_0[f*CC + tid];
    __syncthreads();
    conv_wmma<2>(HB, HA, wf + 0*FRAGS_PER_CONV*FRAG_ELEMS, bias_l, 1,
                 nullptr, xn_l, wd_l, bd_l, lane, wave);
    __syncthreads();

    // ---- blocks 1..3: dil = 2,4,8, identity residual ----
    for (int l = 0; l < 3; ++l) {
        const int dil = 2 << l;
        if (tid < CC) bias_l[tid] = b1_r[((size_t)l*TF + f)*CC + tid];
        __syncthreads();
        conv_wmma<0>(HA, HB, wf + (size_t)(1+2*l)*FRAGS_PER_CONV*FRAG_ELEMS,
                     bias_l, dil, nullptr, xn_l, wd_l, bd_l, lane, wave);
        __syncthreads();
        if (tid < CC) bias_l[tid] = b2_r[((size_t)l*TF + f)*CC + tid];
        __syncthreads();
        conv_wmma<1>(HB, HA, wf + (size_t)(2+2*l)*FRAGS_PER_CONV*FRAG_ELEMS,
                     bias_l, dil, HA, xn_l, wd_l, bd_l, lane, wave);
        __syncthreads();
    }

    // ---- emit feats[b][f*128+co] = last timestep ----
    {
        int co = tid >> 1, bl = tid & 1;
        int b = bg*NBAT + bl;
        int n = bl*CWIN + (CWIN - 1);
        feats[(size_t)b*(TF*CC) + f*CC + co] = bf2f(HA[n*CC + co]);
    }
}

// ---------------------------------------------------------------------------
// Kernel 4: MLP head 1024->256->128->1 + RevIN denorm (tiny tail, 1 WG)
// ---------------------------------------------------------------------------
__global__ __launch_bounds__(256) void tcn_head_kernel(
    const float* __restrict__ feats,
    const float* __restrict__ hw1, const float* __restrict__ hb1,
    const float* __restrict__ hw2, const float* __restrict__ hb2,
    const float* __restrict__ hw3, const float* __restrict__ hb3,
    const float* __restrict__ rw, const float* __restrict__ rb,
    const float* __restrict__ ms, float* __restrict__ out)
{
    __shared__ float h1[32*256];
    __shared__ float h2[32*128];
    const int tid = threadIdx.x;
    for (int r = 0; r < 32; ++r) {
        int e = tid + (r << 8), b = e >> 8, o = e & 255;
        const float* fv = feats + (size_t)b*1024;
        const float* wv = hw1 + (size_t)o*1024;
        float acc = hb1[o];
        for (int k = 0; k < 1024; ++k) acc += fv[k]*wv[k];
        h1[b*256 + o] = acc > 0.f ? acc : 0.01f*acc;
    }
    __syncthreads();
    for (int r = 0; r < 16; ++r) {
        int e = tid + (r << 8), b = e >> 7, o = e & 127;
        const float* hv = h1 + b*256;
        const float* wv = hw2 + (size_t)o*256;
        float acc = hb2[o];
        for (int k = 0; k < 256; ++k) acc += hv[k]*wv[k];
        h2[b*128 + o] = acc > 0.f ? acc : 0.01f*acc;
    }
    __syncthreads();
    if (tid < 32) {
        float acc = hb3[0];
        for (int k = 0; k < 128; ++k) acc += h2[tid*128 + k]*hw3[k];
        float y = (acc - rb[0]) / (rw[0] + 1e-10f);
        out[tid] = y * ms[32 + tid] + ms[tid];
    }
}

// ---------------------------------------------------------------------------
extern "C" void kernel_launch(void* const* d_in, const int* in_sizes, int n_in,
                              void* d_out, int out_size, void* d_ws, size_t ws_size,
                              hipStream_t stream) {
    const float* x       = (const float*)d_in[0];
    const float* revin_w = (const float*)d_in[1];
    const float* revin_b = (const float*)d_in[2];
    const float* w1_0    = (const float*)d_in[3];
    const float* b1_0    = (const float*)d_in[4];
    const float* w2_0    = (const float*)d_in[5];
    const float* b2_0    = (const float*)d_in[6];
    const float* down_w  = (const float*)d_in[7];
    const float* down_b  = (const float*)d_in[8];
    const float* w1_r    = (const float*)d_in[9];
    const float* b1_r    = (const float*)d_in[10];
    const float* w2_r    = (const float*)d_in[11];
    const float* b2_r    = (const float*)d_in[12];
    const float* hw1     = (const float*)d_in[13];
    const float* hb1     = (const float*)d_in[14];
    const float* hw2     = (const float*)d_in[15];
    const float* hb2     = (const float*)d_in[16];
    const float* hw3     = (const float*)d_in[17];
    const float* hb3     = (const float*)d_in[18];
    float* out = (float*)d_out;

    unsigned char* ws = (unsigned char*)d_ws;
    float* xnw  = (float*)ws;                                   // 64 KB
    float* ms   = (float*)(ws + 65536);                         // 256 B
    float* feats = (float*)(ws + 65792);                        // 128 KB
    unsigned short* wswz = (unsigned short*)(ws + 196864);      // ~5.25 MB

    tcn_stats_kernel<<<TB*TF, 128, 0, stream>>>(x, revin_w, revin_b, xnw, ms);

    const int nswz = TF*NCONV_PER_F*FRAGS_PER_CONV*FRAG_ELEMS;  // 2,752,512
    tcn_wswz_kernel<<<nswz/256, 256, 0, stream>>>(w2_0, w1_r, w2_r, wswz);

    const size_t shmem = (size_t)2*NCOL*CC*2 + (NCOL + 3*CC)*sizeof(float); // 66 KB
    tcn_branch_kernel<<<dim3(16, 8), 256, shmem, stream>>>(
        xnw, wswz, w1_0, b1_0, b2_0, down_w, down_b, b1_r, b2_r, feats);

    tcn_head_kernel<<<1, 256, 0, stream>>>(
        feats, hw1, hb1, hw2, hb2, hw3, hb3, revin_w, revin_b, ms, out);
}
